// DistanceLoss_28071906247069
// MI455X (gfx1250) — matrix operand order repaired and checked by reading, not compiled, not run on hardware
//
#include <hip/hip_runtime.h>
#include <hip/hip_bf16.h>

// ---- CDNA5 WMMA types -------------------------------------------------------
typedef __attribute__((ext_vector_type(16))) _Float16 v16h;
typedef __attribute__((ext_vector_type(8)))  float    v8f;

union V16 { uint4 u[2]; v16h v; };

// ---- TDM builtin availability ----------------------------------------------
#if defined(__has_builtin)
#if __has_builtin(__builtin_amdgcn_tensor_load_to_lds) && \
    __has_builtin(__builtin_amdgcn_s_wait_tensorcnt)
#define USE_TDM 1
#endif
#endif

// ---- Problem constants (from reference) ------------------------------------
constexpr int FG  = 16;    // global features
constexpr int FL  = 64;    // local features
constexpr int F   = 80;    // FG + FL
constexpr int D   = 512;   // feature dim
constexpr int WAY = 5;

constexpr int KC     = 128;       // K-chunk (halves) staged in LDS
constexpr int PAD    = KC + 8;    // padded LDS row stride (halves): 272B = 68 dwords
constexpr int NCHUNK = D / KC;    // 4

// ============================================================================
// Kernel 1: L2-normalize each [D]-row of S (global||local) and Q into fp16.
// ============================================================================
__global__ void dl_normalize(const float* __restrict__ sg, const float* __restrict__ sl,
                             const float* __restrict__ qg, const float* __restrict__ ql,
                             _Float16* __restrict__ Sn, _Float16* __restrict__ Qn,
                             int nSrows, int totalRows)
{
    const int wave = threadIdx.x >> 5;
    const int lane = threadIdx.x & 31;
    const int row  = blockIdx.x * (blockDim.x >> 5) + wave;
    if (row >= totalRows) return;

    const float* src;
    _Float16*    dst;
    if (row < nSrows) {
        int item = row / F, f = row % F;
        src = (f < FG) ? sg + ((size_t)item * FG + f) * D
                       : sl + ((size_t)item * FL + (f - FG)) * D;
        dst = Sn + (size_t)row * D;
    } else {
        int r = row - nSrows;
        int item = r / F, f = r % F;
        src = (f < FG) ? qg + ((size_t)item * FG + f) * D
                       : ql + ((size_t)item * FL + (f - FG)) * D;
        dst = Qn + (size_t)r * D;
    }

    const float4* s4 = (const float4*)src;
    float4 v[4];
    float ss = 0.f;
#pragma unroll
    for (int j = 0; j < 4; ++j) {
        v[j] = s4[lane * 4 + j];
        ss += v[j].x * v[j].x + v[j].y * v[j].y + v[j].z * v[j].z + v[j].w * v[j].w;
    }
#pragma unroll
    for (int o = 16; o > 0; o >>= 1) ss += __shfl_xor(ss, o, 32);

    const float sc = 1.f / fmaxf(sqrtf(ss), 1e-12f);

    struct __align__(8) H4 { _Float16 x, y, z, w; };
    H4* d4 = (H4*)dst;
#pragma unroll
    for (int j = 0; j < 4; ++j) {
        H4 h;
        h.x = (_Float16)(v[j].x * sc);
        h.y = (_Float16)(v[j].y * sc);
        h.z = (_Float16)(v[j].z * sc);
        h.w = (_Float16)(v[j].w * sc);
        d4[lane * 4 + j] = h;
    }
}

// ============================================================================
// TDM: issue one 2D tensor_load_to_lds moving an 80-row x 256B tile
// (row stride 512 halves in global) into LDS with +16B/row hardware padding.
// ============================================================================
#if USE_TDM
typedef unsigned int tdm_v4u __attribute__((ext_vector_type(4)));
typedef int          tdm_v8i __attribute__((ext_vector_type(8)));
typedef int          tdm_v4i __attribute__((ext_vector_type(4)));

__device__ __forceinline__ void tdm_load_chunk(unsigned int lds_addr, const _Float16* gptr)
{
    const unsigned long long ga = (unsigned long long)(uintptr_t)gptr;

    tdm_v4u g0;
    g0.x = 1u;                                   // count=1, user descriptor
    g0.y = lds_addr;                             // LDS byte address
    g0.z = (unsigned int)ga;                     // global_addr[31:0]
    g0.w = (unsigned int)((ga >> 32) & 0x01FFFFFFu) | 0x80000000u;  // addr[56:32] | type=2

    tdm_v8i g1;
    g1[0] = (int)((1u << 16)      // data_size = 2 bytes
                | (1u << 20)      // pad_enable
                | (5u << 22)      // pad_interval: 64 dwords (=256B, one row)
                | (3u << 25));    // pad_amount:   4 dwords (=16B)  -> PAD stride
    g1[1] = (int)(512u << 16);    // tensor_dim0 = 512 (bits 79:48, low half here)
    g1[2] = (int)(80u  << 16);    // tensor_dim0 hi=0 | tensor_dim1 = 80
    g1[3] = (int)(128u << 16);    // tensor_dim1 hi=0 | tile_dim0 = 128
    g1[4] = (int)80u;             // tile_dim1 = 80, tile_dim2 = 0
    g1[5] = (int)512u;            // tensor_dim0_stride = 512 (low 32)
    g1[6] = 0;                    // stride0 hi | tensor_dim1_stride lo
    g1[7] = 0;

    const tdm_v4i z4 = (tdm_v4i)0;
#if defined(__clang_major__) && (__clang_major__ >= 23)
    const tdm_v8i z8 = (tdm_v8i)0;
    __builtin_amdgcn_tensor_load_to_lds(g0, g1, z4, z4, z8, 0);
#else
    __builtin_amdgcn_tensor_load_to_lds(g0, g1, z4, z4, 0);
#endif
}
#endif // USE_TDM

// ---- WMMA compute for one staged K-chunk (KC=128) ---------------------------
__device__ __forceinline__ void dl_compute_chunk(const _Float16* __restrict__ bS,
                                                 const _Float16* __restrict__ bQ,
                                                 int wave, int mrow, bool hi, v8f acc[5])
{
    const _Float16* arow = bS + (wave * 16 + mrow) * PAD;
#pragma unroll
    for (int ks = 0; ks < KC / 32; ++ks) {
        const int k0 = ks * 32;
        // A fragment (16x32 f16): lanes 0-15 K {k0..+7, k0+16..+23}; hi lanes +8
        const int ka = k0 + (hi ? 8 : 0);
        V16 A;
        A.u[0] = *(const uint4*)(arow + ka);
        A.u[1] = *(const uint4*)(arow + ka + 16);
        // B fragments (32x16 f16): lanes 0-15 K {k0..+7, k0+16..+23}; hi lanes +8
        const int kb = k0 + (hi ? 8 : 0);
        V16 Bf[5];
#pragma unroll
        for (int nt = 0; nt < 5; ++nt) {
            const _Float16* brow = bQ + (nt * 16 + mrow) * PAD + kb;
            Bf[nt].u[0] = *(const uint4*)(brow);
            Bf[nt].u[1] = *(const uint4*)(brow + 16);
        }
#pragma unroll
        for (int nt = 0; nt < 5; ++nt)
            acc[nt] = __builtin_amdgcn_wmma_f32_16x16x32_f16(
                false, A.v, false, Bf[nt].v, (short)0, acc[nt], false, false);
    }
}

// ============================================================================
// Kernel 2: per (support s, query q): fro2 = sum_{80x80} (1 - cos_sim)^2.
// 5 waves/block; TDM double-buffered K-chunks when available.
// ============================================================================
__global__ __launch_bounds__(160) void dl_pairdist(const _Float16* __restrict__ Sn,
                                                   const _Float16* __restrict__ Qn,
                                                   float* __restrict__ cum, int nQ)
{
    __shared__ __align__(16) _Float16 sS[2][F * PAD];
    __shared__ __align__(16) _Float16 sQ[2][F * PAD];
    __shared__ float wsum[5];

    const int s = blockIdx.x / nQ;
    const int q = blockIdx.x % nQ;

    const int tid  = threadIdx.x;
    const int lane = tid & 31;
    const int wave = tid >> 5;          // 0..4  -> S strip
    const int mrow = lane & 15;
    const bool hi  = lane >= 16;

    const _Float16* Srow0 = Sn + (size_t)s * F * D;
    const _Float16* Qrow0 = Qn + (size_t)q * F * D;

    v8f acc[5];
#pragma unroll
    for (int t = 0; t < 5; ++t) acc[t] = (v8f){0.f,0.f,0.f,0.f,0.f,0.f,0.f,0.f};

#if USE_TDM
    // LDS byte addresses of the two buffers (generic addr low 32 bits == LDS offset)
    const unsigned int ldsS0 = (unsigned int)(uintptr_t)&sS[0][0];
    const unsigned int ldsS1 = (unsigned int)(uintptr_t)&sS[1][0];
    const unsigned int ldsQ0 = (unsigned int)(uintptr_t)&sQ[0][0];
    const unsigned int ldsQ1 = (unsigned int)(uintptr_t)&sQ[1][0];

    if (wave == 0) {                       // prologue: chunk 0 into buffer 0
        tdm_load_chunk(ldsS0, Srow0);
        tdm_load_chunk(ldsQ0, Qrow0);
    }
    for (int kc = 0; kc < NCHUNK; ++kc) {
        const int buf = kc & 1;
        if (wave == 0) {
            if (kc + 1 < NCHUNK) {         // kick off next chunk into other buffer
                const int nb = buf ^ 1;
                tdm_load_chunk(nb ? ldsS1 : ldsS0, Srow0 + (kc + 1) * KC);
                tdm_load_chunk(nb ? ldsQ1 : ldsQ0, Qrow0 + (kc + 1) * KC);
                __builtin_amdgcn_s_wait_tensorcnt(2);  // current chunk's pair done
            } else {
                __builtin_amdgcn_s_wait_tensorcnt(0);
            }
        }
        __syncthreads();                   // data visible to all 5 waves
        dl_compute_chunk(sS[buf], sQ[buf], wave, mrow, hi, acc);
        __syncthreads();                   // buffer reusable before next TDM issue
    }
#else
    // Fallback: synchronous staged copy (single buffer)
    for (int kc = 0; kc < NCHUNK; ++kc) {
        const int kbase = kc * KC;
        __syncthreads();
        for (int e = tid; e < F * (KC / 8); e += 160) {   // 80 rows x 16 uint4
            const int row = e >> 4, c = e & 15;
            *(uint4*)&sS[0][row * PAD + c * 8] =
                *(const uint4*)(Srow0 + (size_t)row * D + kbase + c * 8);
            *(uint4*)&sQ[0][row * PAD + c * 8] =
                *(const uint4*)(Qrow0 + (size_t)row * D + kbase + c * 8);
            if (kc + 1 < NCHUNK) {
                __builtin_prefetch(Srow0 + (size_t)row * D + kbase + KC + c * 8, 0, 1);
                __builtin_prefetch(Qrow0 + (size_t)row * D + kbase + KC + c * 8, 0, 1);
            }
        }
        __syncthreads();
        dl_compute_chunk(sS[0], sQ[0], wave, mrow, hi, acc);
    }
#endif

    // sum (1 - sim)^2 over this wave's 5 tiles
    float loc = 0.f;
#pragma unroll
    for (int nt = 0; nt < 5; ++nt)
#pragma unroll
        for (int e = 0; e < 8; ++e) {
            const float dv = 1.f - acc[nt][e];
            loc += dv * dv;
        }
#pragma unroll
    for (int o = 16; o > 0; o >>= 1) loc += __shfl_xor(loc, o, 32);
    if (lane == 0) wsum[wave] = loc;
    __syncthreads();
    if (tid == 0)
        cum[(size_t)s * nQ + q] = wsum[0] + wsum[1] + wsum[2] + wsum[3] + wsum[4];
}

// ============================================================================
// Kernel 3: logits[q][c] = -(2/count_c) * sum_{s: label==c} fro2[s][q]
// ============================================================================
__global__ void dl_logits(const float* __restrict__ cum, const int* __restrict__ labels,
                          float* __restrict__ out, int nS, int nQ)
{
    const int idx = blockIdx.x * blockDim.x + threadIdx.x;
    if (idx >= nQ * WAY) return;
    const int q = idx / WAY, c = idx % WAY;
    float sum = 0.f;
    int cnt = 0;
    for (int si = 0; si < nS; ++si)
        if (labels[si] == c) { sum += cum[(size_t)si * nQ + q]; ++cnt; }
    out[(size_t)q * WAY + c] = -(2.f * sum) / (float)(cnt > 0 ? cnt : 1);
}

// ============================================================================
static inline size_t alignUp(size_t x, size_t a) { return (x + a - 1) & ~(a - 1); }

extern "C" void kernel_launch(void* const* d_in, const int* in_sizes, int n_in,
                              void* d_out, int out_size, void* d_ws, size_t ws_size,
                              hipStream_t stream)
{
    const float* sg     = (const float*)d_in[0];
    const float* sl     = (const float*)d_in[1];
    const int*   labels = (const int*)d_in[2];
    const float* qg     = (const float*)d_in[3];
    const float* ql     = (const float*)d_in[4];

    const int nS = in_sizes[2];               // 25
    const int nQ = in_sizes[3] / (FG * D);    // 200

    const size_t snB = alignUp((size_t)nS * F * D * sizeof(_Float16), 256);
    const size_t qnB = alignUp((size_t)nQ * F * D * sizeof(_Float16), 256);

    _Float16* Sn  = (_Float16*)d_ws;
    _Float16* Qn  = (_Float16*)((char*)d_ws + snB);
    float*    cum = (float*)((char*)d_ws + snB + qnB);

    // Pass 1: normalize -> fp16
    const int totalRows = (nS + nQ) * F;      // 18000 rows
    const int wavesPerBlock = 8;              // 256 threads
    dl_normalize<<<(totalRows + wavesPerBlock - 1) / wavesPerBlock, 32 * wavesPerBlock,
                   0, stream>>>(sg, sl, qg, ql, Sn, Qn, nS * F, totalRows);

    // Pass 2: fused WMMA block-GEMM + Frobenius reduction, one block per (s,q)
    dl_pairdist<<<nS * nQ, 160, 0, stream>>>(Sn, Qn, cum, nQ);

    // Pass 3: class-mean logits
    const int tot = nQ * WAY;
    dl_logits<<<(tot + 255) / 256, 256, 0, stream>>>(cum, labels, (float*)d_out, nS, nQ);
}
